// MoELayerStacks_17437567221959
// MI455X (gfx1250) — matrix operand order repaired and checked by reading, not compiled, not run on hardware
//
#include <hip/hip_runtime.h>
#include <math.h>

// ---------------------------------------------------------------------------
// MoE layer stack for MI455X (gfx1250, wave32, WMMA).
// Phase 1: x(32768x1024) x w1^T(8 x 16 x 1024) via v_wmma_f32_16x16x32_bf16,
//          streaming x from HBM exactly once (128 MB -> ~5.5us floor).
// Phase 2: layer-2 (30->32) ALSO on WMMA (K padded 30->32), so the tiny MLP
//          no longer dominates as scalar VALU work; layer-3 (dot-32) done with
//          a 4-step shfl_xor cross-lane reduction.
// ---------------------------------------------------------------------------

typedef __bf16 v16bf __attribute__((ext_vector_type(16)));
typedef float  v8f   __attribute__((ext_vector_type(8)));

union Frag { uint4 q[2]; v16bf v; };

#define NE       8
#define BTOK     32768
#define D_L1     1024
#define TILE_TOK 64      // tokens per workgroup
#define KCHUNK   64      // K elements staged in LDS per iteration

static __device__ __forceinline__ unsigned short bf16_bits(float f) {
    return __builtin_bit_cast(unsigned short, (__bf16)f);
}

// ---------------------------------------------------------------------------
// Kernel 0: zero the 32-float accumulator block in workspace (every launch).
// ---------------------------------------------------------------------------
__global__ __launch_bounds__(32) void zero_acc_kernel(float* __restrict__ acc) {
    acc[threadIdx.x] = 0.0f;
}

// ---------------------------------------------------------------------------
// Kernel 1: fuse w1 = l1_w + tile(l1_fw) and convert to bf16.
// ---------------------------------------------------------------------------
__global__ __launch_bounds__(256) void wprep_kernel(const float* __restrict__ l1_w,
                                                    const float* __restrict__ l1_fw,
                                                    unsigned short* __restrict__ w1b) {
    int i   = blockIdx.x * 256 + threadIdx.x;   // 0 .. 131071
    int row = i >> 10;                          // e*16 + o
    int k   = i & 1023;
    float v = l1_w[i] + l1_fw[(row & 15) * 1024 + k];
    w1b[i]  = bf16_bits(v);
}

// ---------------------------------------------------------------------------
// Kernel 2: router. One thread per token; LDS-then-global f32 atomics.
// acc layout: [0]=target_prob_sum [1]=agree_sum [2]=tce_sum [3]=z_sum
//             [4..11]=frac counts  [12..19]=gate prob sums
// ---------------------------------------------------------------------------
__global__ __launch_bounds__(256) void router_kernel(const float* __restrict__ x,
                                                     const int*   __restrict__ ls,
                                                     const float* __restrict__ rw,
                                                     const float* __restrict__ rb,
                                                     int*   __restrict__ idx,
                                                     float* __restrict__ acc) {
    __shared__ float rws[NE * 64];
    __shared__ float rbs[NE];
    __shared__ float sacc[20];

    int tid = threadIdx.x;
    for (int i = tid; i < NE * 64; i += 256) rws[i] = rw[i];
    if (tid < NE) rbs[tid] = rb[tid];
    if (tid < 20) sacc[tid] = 0.0f;
    __syncthreads();

    int b = blockIdx.x * 256 + tid;
    const float* xb = x + (size_t)b * D_L1;

    float ri[64];
    #pragma unroll
    for (int j = 0; j < 32; j += 4) {
        float4 f = *(const float4*)(xb + j);
        ri[j] = f.x; ri[j + 1] = f.y; ri[j + 2] = f.z; ri[j + 3] = f.w;
    }
    #pragma unroll
    for (int j = 0; j < 32; j += 4) {
        float4 f = *(const float4*)(xb + 512 + j);
        ri[32 + j] = f.x; ri[33 + j] = f.y; ri[34 + j] = f.z; ri[35 + j] = f.w;
    }

    float lg[NE];
    #pragma unroll
    for (int e = 0; e < NE; ++e) {
        float s = rbs[e];
        #pragma unroll
        for (int j = 0; j < 64; ++j) s = fmaf(ri[j], rws[e * 64 + j], s);
        lg[e] = s;
    }

    float mx = lg[0];
    #pragma unroll
    for (int e = 1; e < NE; ++e) mx = fmaxf(mx, lg[e]);
    float se = 0.0f;
    #pragma unroll
    for (int e = 0; e < NE; ++e) se += expf(lg[e] - mx);
    float lse = mx + logf(se);

    int   bi = 0; float best = lg[0];
    #pragma unroll
    for (int e = 1; e < NE; ++e) { if (lg[e] > best) { best = lg[e]; bi = e; } }

    int lsi = ls[b];
    float logp_ls = lg[lsi] - lse;

    idx[b] = bi;
    atomicAdd(&sacc[0], expf(logp_ls));                 // target_prob
    atomicAdd(&sacc[1], (bi == lsi) ? 1.0f : 0.0f);     // agreement
    atomicAdd(&sacc[2], -logp_ls);                      // teacher CE
    atomicAdd(&sacc[3], lse * lse);                     // z loss
    atomicAdd(&sacc[4 + bi], 1.0f);                     // fraction counts
    #pragma unroll
    for (int e = 0; e < NE; ++e) atomicAdd(&sacc[12 + e], expf(lg[e] - lse));

    __syncthreads();
    if (tid < 20) atomicAdd(&acc[tid], sacc[tid]);
}

// ---------------------------------------------------------------------------
// Kernel 3: main expert stack. 256 threads (8 waves) per 64 tokens.
// Wave w == expert w throughout.
// ---------------------------------------------------------------------------
__global__ __launch_bounds__(256) void moe_main_kernel(const float* __restrict__ x,
                                                       const unsigned short* __restrict__ w1b,
                                                       const float* __restrict__ l1_b,
                                                       const float* __restrict__ l1_fb,
                                                       const float* __restrict__ l2_w,
                                                       const float* __restrict__ l2_b,
                                                       const float* __restrict__ out_w,
                                                       const float* __restrict__ out_b,
                                                       const int*   __restrict__ idx,
                                                       float* __restrict__ out) {
    __shared__ __bf16 xs[TILE_TOK * KCHUNK];          // 8 KB  x chunk (bf16)
    __shared__ __bf16 wch[NE * 16 * KCHUNK];          // 16 KB w1 chunk, all experts
    __shared__ float  l1s[TILE_TOK * NE * 16];        // 32 KB l1c staging
    __shared__ __bf16 acts[NE * TILE_TOK * 32];       // 32 KB act rows (30+2 pad)
    __shared__ __bf16 w2b[NE * 32 * 32];              // 16 KB w2 bf16 (30+2 pad)
    __shared__ float  xoutس[TILE_TOK * NE];           // placeholder - renamed below
    __shared__ float  w3s[NE * 32];
    __shared__ float  b2s[NE * 32];
    __shared__ float  b3s[NE];
    __shared__ float  b1s[NE * 16];

    float* xout = xoutس;  // l1x_out staging (2 KB)

    const int tid  = threadIdx.x;
    const int tok0 = blockIdx.x * TILE_TOK;
    const int lane = tid & 31;
    const int wv   = tid >> 5;           // expert id for this wave
    const int h    = lane >> 4;          // half-lane group
    const int mn   = lane & 15;          // M (A frag) / N (B frag) index

    // ---- Preload small operands ----
    {   // w2 -> bf16, 32-wide rows (pad k=30,31 with zeros). Row per thread.
        const float* src = l2_w + tid * 30;
        __bf16* dst = &w2b[tid * 32];
        #pragma unroll
        for (int j = 0; j < 30; ++j) dst[j] = (__bf16)src[j];
        dst[30] = (__bf16)0.0f; dst[31] = (__bf16)0.0f;
    }
    if (tid < NE * 32) { w3s[tid] = out_w[tid]; b2s[tid] = l2_b[tid]; }
    if (tid < NE)      b3s[tid] = out_b[tid];
    if (tid < NE * 16) b1s[tid] = l1_b[tid] + l1_fb[tid & 15];

    // ---- Phase 1: l1c = x @ w1^T over K=1024, WMMA bf16 ----
    v8f acc[4] = {};

    for (int kc = 0; kc < D_L1 / KCHUNK; ++kc) {
        const int k0 = kc * KCHUNK;
        __syncthreads();   // previous chunk fully consumed (also covers preload)

        {   // stage x chunk: thread -> token tid>>2, quarter tid&3
            int t = tid >> 2, q = tid & 3;
            const float* src = x + (size_t)(tok0 + t) * D_L1 + k0 + q * 16;
            __bf16* dst = &xs[t * KCHUNK + q * 16];
            #pragma unroll
            for (int j = 0; j < 16; j += 4) {
                float4 f = *(const float4*)(src + j);
                dst[j + 0] = (__bf16)f.x; dst[j + 1] = (__bf16)f.y;
                dst[j + 2] = (__bf16)f.z; dst[j + 3] = (__bf16)f.w;
            }
        }
        {   // stage w1 chunk: 128 rows x 64 bf16 == 1024 uint4, 4 per thread
            #pragma unroll
            for (int j = 0; j < 4; ++j) {
                int g   = tid * 4 + j;
                int row = g >> 3;
                int c4  = g & 7;
                ((uint4*)&wch[row * KCHUNK])[c4] =
                    ((const uint4*)(w1b + row * D_L1 + k0))[c4];
            }
        }
        __syncthreads();

        #pragma unroll
        for (int T = 0; T < 4; ++T) {
            #pragma unroll
            for (int s = 0; s < 2; ++s) {
                const int ks = s * 32;
                Frag af, bf_;
                const __bf16* xrow = &xs[(T * 16 + mn) * KCHUNK + ks];
                af.q[0] = *(const uint4*)(xrow + 16 * h);
                af.q[1] = *(const uint4*)(xrow + 16 * h + 8);
                const __bf16* wrow = &wch[(wv * 16 + mn) * KCHUNK + ks + 16 * h];
                bf_.q[0] = *(const uint4*)(wrow);
                bf_.q[1] = *(const uint4*)(wrow + 8);
                acc[T] = __builtin_amdgcn_wmma_f32_16x16x32_bf16(
                    false, af.v, false, bf_.v, (short)0, acc[T], false, false);
            }
        }
    }

    // ---- Stage l1c (+bias) to LDS. lane -> o = mn, VGPR r -> m = r + 8h ----
    __syncthreads();
    #pragma unroll
    for (int T = 0; T < 4; ++T) {
        const int mbase = T * 16 + 8 * h;
        #pragma unroll
        for (int r = 0; r < 8; ++r) {
            l1s[(mbase + r) * (NE * 16) + wv * 16 + mn] = acc[T][r] + b1s[wv * 16 + mn];
        }
    }
    __syncthreads();

    // ---- Phase 1.5: build bf16 act rows (32-wide, k=30,31 zero) + xout ----
    #pragma unroll
    for (int pp = 0; pp < 2; ++pp) {
        const int p  = tid + pp * 256;
        const int tl = p >> 3;
        const int ee = p & 7;
        const float* lv = &l1s[tl * (NE * 16) + ee * 16];

        float a[32];
        #pragma unroll
        for (int j = 0; j < 15; ++j) {
            float v  = lv[j];
            a[j]      = fminf(v * v * (255.0f / 256.0f), 1.0f);   // >= 0 already
            a[15 + j] = fminf(fmaxf(v, 0.0f), 1.0f);
        }
        a[30] = 0.0f; a[31] = 0.0f;

        unsigned int pk[16];
        #pragma unroll
        for (int w = 0; w < 16; ++w)
            pk[w] = (unsigned int)bf16_bits(a[2 * w]) |
                    ((unsigned int)bf16_bits(a[2 * w + 1]) << 16);

        uint4* dst = (uint4*)&acts[(ee * TILE_TOK + tl) * 32];
        dst[0] = ((uint4*)pk)[0];
        dst[1] = ((uint4*)pk)[1];
        dst[2] = ((uint4*)pk)[2];
        dst[3] = ((uint4*)pk)[3];

        xout[tl * NE + ee] = lv[15];
    }
    __syncthreads();

    // ---- Phase B: layer 2 on WMMA (K=32), layer 3 via shfl reduction ----
    // B fragments (w2 for this expert), hoisted: n2 = 0,1 -> o2 = n2*16+mn.
    Frag bf0, bf1;
    {
        const __bf16* w0 = &w2b[(wv * 32 + 0 * 16 + mn) * 32 + 16 * h];
        bf0.q[0] = *(const uint4*)(w0);
        bf0.q[1] = *(const uint4*)(w0 + 8);
        const __bf16* w1r = &w2b[(wv * 32 + 1 * 16 + mn) * 32 + 16 * h];
        bf1.q[0] = *(const uint4*)(w1r);
        bf1.q[1] = *(const uint4*)(w1r + 8);
    }
    const float b2v0 = b2s[wv * 32 + mn];
    const float b2v1 = b2s[wv * 32 + 16 + mn];
    const float w3v0 = w3s[wv * 32 + mn];
    const float w3v1 = w3s[wv * 32 + 16 + mn];
    const float b3v  = b3s[wv];

    #pragma unroll
    for (int T = 0; T < 4; ++T) {
        Frag af;
        const __bf16* arow = &acts[(wv * TILE_TOK + T * 16 + mn) * 32];
        af.q[0] = *(const uint4*)(arow + 16 * h);
        af.q[1] = *(const uint4*)(arow + 16 * h + 8);

        v8f c0 = {}, c1 = {};
        c0 = __builtin_amdgcn_wmma_f32_16x16x32_bf16(
                 false, af.v, false, bf0.v, (short)0, c0, false, false);
        c1 = __builtin_amdgcn_wmma_f32_16x16x32_bf16(
                 false, af.v, false, bf1.v, (short)0, c1, false, false);

        float rc[8];
        #pragma unroll
        for (int r = 0; r < 8; ++r) {
            float v0 = fminf(fmaxf(c0[r] + b2v0, 0.0f), 1.0f);
            float v1 = fminf(fmaxf(c1[r] + b2v1, 0.0f), 1.0f);
            rc[r] = fmaf(v0, w3v0, v1 * w3v1);
        }
        // Sum over the 16 lanes of each half (o2 dimension).
        #pragma unroll
        for (int r = 0; r < 8; ++r) {
            rc[r] += __shfl_xor(rc[r], 1);
            rc[r] += __shfl_xor(rc[r], 2);
            rc[r] += __shfl_xor(rc[r], 4);
            rc[r] += __shfl_xor(rc[r], 8);
        }
        if (mn == 0) {
            #pragma unroll
            for (int r = 0; r < 8; ++r) {
                const int mloc = T * 16 + 8 * h + r;
                const int tok  = tok0 + mloc;
                float val = rc[r] + b3v + xout[mloc * NE + wv];
                if (idx[tok] == wv) out[tok] = val;
            }
        }
    }
}

// ---------------------------------------------------------------------------
// Kernel 4: finalize scalar outputs.
// ---------------------------------------------------------------------------
__global__ __launch_bounds__(32) void finalize_kernel(const float* __restrict__ acc,
                                                      float* __restrict__ out) {
    if (threadIdx.x != 0) return;
    const float invB = 1.0f / (float)BTOK;
    float tp  = acc[0] * invB;
    float ag  = acc[1] * invB;
    float tce = acc[2] * invB;
    float z   = acc[3] * invB;
    float aux = 0.0f;
    float frac[NE], avg[NE];
    for (int e = 0; e < NE; ++e) {
        frac[e] = acc[4 + e]  * invB;
        avg[e]  = acc[12 + e] * invB;
        float fl = fmaxf(0.05f - frac[e], 0.0f);
        float cp = fmaxf(frac[e] - 0.5f, 0.0f);
        aux += fl * fl + cp * cp;
    }
    aux *= (1.0f / (float)NE);
    float rl = 0.01f * aux + 0.001f * z + 0.5f * tce;
    out[BTOK + 0] = rl;
    out[BTOK + 1] = aux;
    out[BTOK + 2] = z;
    out[BTOK + 3] = tce;
    for (int e = 0; e < NE; ++e) out[BTOK + 4 + e]  = frac[e];
    for (int e = 0; e < NE; ++e) out[BTOK + 12 + e] = avg[e];
    out[BTOK + 20] = tp;
    out[BTOK + 21] = ag;
}

// ---------------------------------------------------------------------------
// Launch: zero accumulators -> weight prep -> router -> WMMA experts -> scalars
// Workspace layout: [0,128) accumulators, [128, 128+131072) idx int32,
//                   [131200, 131200+262144) w1 bf16.
// ---------------------------------------------------------------------------
extern "C" void kernel_launch(void* const* d_in, const int* in_sizes, int n_in,
                              void* d_out, int out_size, void* d_ws, size_t ws_size,
                              hipStream_t stream) {
    const float* x     = (const float*)d_in[0];
    const int*   ls    = (const int*)  d_in[1];
    const float* rw    = (const float*)d_in[2];
    const float* rb    = (const float*)d_in[3];
    const float* l1_w  = (const float*)d_in[4];
    const float* l1_fw = (const float*)d_in[5];
    const float* l1_b  = (const float*)d_in[6];
    const float* l1_fb = (const float*)d_in[7];
    const float* l2_w  = (const float*)d_in[8];
    const float* l2_b  = (const float*)d_in[9];
    const float* out_w = (const float*)d_in[10];
    const float* out_b = (const float*)d_in[11];
    float* out = (float*)d_out;

    float*          acc = (float*)d_ws;
    int*            idx = (int*)((char*)d_ws + 128);
    unsigned short* w1b = (unsigned short*)((char*)d_ws + 128 + BTOK * sizeof(int));

    zero_acc_kernel<<<1, 32, 0, stream>>>(acc);
    wprep_kernel<<<(NE * 16 * D_L1) / 256, 256, 0, stream>>>(l1_w, l1_fw, w1b);
    router_kernel<<<BTOK / 256, 256, 0, stream>>>(x, ls, rw, rb, idx, acc);
    moe_main_kernel<<<BTOK / TILE_TOK, 256, 0, stream>>>(x, w1b, l1_b, l1_fb,
                                                         l2_w, l2_b, out_w, out_b,
                                                         idx, out);
    finalize_kernel<<<1, 32, 0, stream>>>(acc, out);
}
